// LightGraphQA_88287347736844
// MI455X (gfx1250) — matrix operand order repaired and checked by reading, not compiled, not run on hardware
//
#include <hip/hip_runtime.h>
#include <hip/hip_bf16.h>

// ---------------------------------------------------------------------------
// LightGraphQA interaction network, T=16 steps.
// fp16 activations/weights, fp32 accumulation via v_wmma_f32_16x16x32_f16
// (CDNA5 wave32 WMMA). Weights pre-transposed to column-major fp16 so each
// lane's B-fragment is two 8-half contiguous runs. Wave tile 32(M) x 64(N): each
// B fragment feeds two WMMAs, halving LDS traffic per matrix op.
// ---------------------------------------------------------------------------

typedef __attribute__((ext_vector_type(16))) _Float16 v16h;
typedef __attribute__((ext_vector_type(8)))  _Float16 v8h;
typedef __attribute__((ext_vector_type(8)))  float    v8f;

#define B_SZ   1024
#define N_OBJ  6
#define D_DIM  128
#define N_REL  30
#define T_STEP 16

__device__ __forceinline__ v16h frag16(const _Float16* p0, const _Float16* p1) {
    v8h lo = *(const v8h*)p0;
    v8h hi = *(const v8h*)p1;
    return __builtin_shufflevector(lo, hi, 0,1,2,3,4,5,6,7,8,9,10,11,12,13,14,15);
}

// ---------------------------------------------------------------------------
// Generic fused GEMM: Out[M,N] = act(A[M,K] @ W[K,N] + bias[N])
// A: fp16 row-major. Wt: fp16 column-major (Wt[n*K+k]). Out: fp16 row-major.
// Block: 256 threads (8 wave32). Block tile 256(M) x 64(N). Wave tile 32x64.
// M must be a multiple of 256 (30720 and 6144 both are). N multiple of 64.
// ---------------------------------------------------------------------------
template<int K, int N, bool RELU>
__global__ __launch_bounds__(256)
void gemm_bias_act(const _Float16* __restrict__ A,
                   const _Float16* __restrict__ Wt,
                   const float*    __restrict__ bias,
                   _Float16*       __restrict__ Out)
{
    constexpr int NT = 64;                 // N-tile per block
    __shared__ _Float16 sW[NT * K];        // <= 64*384*2 = 48KB

    const int wave    = threadIdx.x >> 5;
    const int lane    = threadIdx.x & 31;
    const int rowBase = blockIdx.x * 256 + wave * 32;
    const int colBase = blockIdx.y * NT;

    // Cooperative stage of the Wt column slice into LDS (vector b128 copies).
    {
        const _Float16* src = Wt + (size_t)colBase * K;
        const int total = (NT * K) / 8;
        for (int i = threadIdx.x; i < total; i += 256)
            *(v8h*)&sW[i * 8] = *(const v8h*)&src[i * 8];
    }
    __syncthreads();

    v8f acc0[NT / 16] = {};                // rows [rowBase,    +16)
    v8f acc1[NT / 16] = {};                // rows [rowBase+16, +16)

    const int r  = lane & 15;              // row-in-tile / col-in-tile index
    const int sA = (lane >> 4) * 8;        // A frag K-offset split (ISA layout)
    const int sB = (lane >> 4) * 16;       // unused (B frag uses sA)
    const _Float16* arow0 = A + (size_t)(rowBase + r) * K;
    const _Float16* arow1 = A + (size_t)(rowBase + 16 + r) * K;

    for (int k0 = 0; k0 < K; k0 += 32) {
        // A fragments 16x32 fp16: lanes 0-15 hold K {0..7,16..23}, lanes
        // 16-31 hold K {8..15,24..31} of row (lane&15).
        v16h a0 = frag16(&arow0[k0 + sA], &arow0[k0 + sA + 16]);
        v16h a1 = frag16(&arow1[k0 + sA], &arow1[k0 + sA + 16]);

        // Load ALL B fragments for this K-chunk first so the scheduler can
        // overlap ds_loads with the WMMA burst (partial s_wait_dscnt).
        v16h bf[NT / 16];
#pragma unroll
        for (int t = 0; t < NT / 16; ++t) {
            // B fragment 32x16 fp16: lane = column (t*16 + r), lanes 0-15
            // hold K {0..7,16..23}, lanes 16-31 hold K {8..15,24..31}
            // (same K order as the A fragment) in the column-major weight tile.
            const _Float16* bcol = &sW[(size_t)(t * 16 + r) * K + k0 + sA];
            bf[t] = frag16(bcol, bcol + 16);
        }
#pragma unroll
        for (int t = 0; t < NT / 16; ++t) {
            acc0[t] = __builtin_amdgcn_wmma_f32_16x16x32_f16(
                false, a0, false, bf[t], (short)0, acc0[t], false, false);
            acc1[t] = __builtin_amdgcn_wmma_f32_16x16x32_f16(
                false, a1, false, bf[t], (short)0, acc1[t], false, false);
        }
    }

    // Epilogue: C/D layout -> VGPR v: lanes 0-15 row v, lanes 16-31 row v+8.
    const int rOff = (lane >> 4) * 8;
#pragma unroll
    for (int t = 0; t < NT / 16; ++t) {
        const int col = colBase + t * 16 + r;
        const float bs = bias[col];
#pragma unroll
        for (int v = 0; v < 8; ++v) {
            float x0 = acc0[t][v] + bs;
            float x1 = acc1[t][v] + bs;
            if (RELU) { x0 = fmaxf(x0, 0.0f); x1 = fmaxf(x1, 0.0f); }
            Out[(size_t)(rowBase + v + rOff) * N + col]      = (_Float16)x0;
            Out[(size_t)(rowBase + 16 + v + rOff) * N + col] = (_Float16)x1;
        }
    }
}

// ---------------------------------------------------------------------------
// Weight prep: W[K,N] fp32 row-major -> Wt[N,K] fp16 column-major.
// ---------------------------------------------------------------------------
__global__ void transpose_cvt(const float* __restrict__ W, _Float16* __restrict__ Wt,
                              int K, int N)
{
    int idx = blockIdx.x * blockDim.x + threadIdx.x;
    if (idx >= K * N) return;
    int k = idx / N, n = idx % N;
    Wt[(size_t)n * K + k] = (_Float16)W[(size_t)k * N + n];
}

__global__ void cvt_f32_f16(const float* __restrict__ in, _Float16* __restrict__ out, int n)
{
    int i = blockIdx.x * blockDim.x + threadIdx.x;
    if (i < n) out[i] = (_Float16)in[i];
}

__global__ void cvt_f16_f32(const _Float16* __restrict__ in, float* __restrict__ out, int n)
{
    int i = blockIdx.x * blockDim.x + threadIdx.x;
    if (i < n) out[i] = (float)in[i];
}

// ---------------------------------------------------------------------------
// Edge gather: e[b,r,:] = [ nd[b,send[r],:] | nd[b,recv[r],:] ]  (fp16 copy)
// Incidence is the fixed fully-connected graph: r enumerates (i,j), i!=j,
// i-major: send=i, recv=j -> computed analytically (Rs/Rr inputs unused).
// ---------------------------------------------------------------------------
__global__ void gather_e(const _Float16* __restrict__ nd, _Float16* __restrict__ e)
{
    int idx = blockIdx.x * blockDim.x + threadIdx.x;        // [0, B*30*32)
    if (idx >= B_SZ * N_REL * 32) return;
    int row = idx >> 5;                                     // edge row
    int c   = idx & 31;                                     // 8-half chunk
    int b   = row / N_REL;
    int r   = row % N_REL;
    int i   = r / 5;
    int jj  = r % 5;
    int j   = jj + (jj >= i ? 1 : 0);
    int d   = c * 8;
    int node = (d < D_DIM) ? i : j;
    int off  = (d < D_DIM) ? d : (d - D_DIM);
    *(v8h*)&e[(size_t)row * 256 + d] =
        *(const v8h*)&nd[((size_t)b * N_OBJ + node) * D_DIM + off];
}

// ---------------------------------------------------------------------------
// u[b,n,:] = [ nd[b,n,:] | sum_{incoming r} eff[b,r,:] | black[b,:] ]  fp16
// Incoming edges of node n: r = i*5 + (n<i ? n : n-1) for all i != n.
// ---------------------------------------------------------------------------
__global__ void build_u(const _Float16* __restrict__ nd,
                        const _Float16* __restrict__ eff,
                        const _Float16* __restrict__ blk,
                        _Float16* __restrict__ u)
{
    int row = blockIdx.x;                 // [0, B*6)
    int d   = threadIdx.x;                // [0, 128)
    int b   = row / N_OBJ;
    int n   = row % N_OBJ;
    _Float16* ur = u + (size_t)row * 384;
    ur[d] = nd[(size_t)row * D_DIM + d];
    float s = 0.0f;
#pragma unroll
    for (int i = 0; i < N_OBJ; ++i) {
        if (i == n) continue;
        int r = i * 5 + (n < i ? n : n - 1);
        s += (float)eff[((size_t)b * N_REL + r) * D_DIM + d];
    }
    ur[128 + d] = (_Float16)s;
    ur[256 + d] = blk[(size_t)b * D_DIM + d];
}

// ---------------------------------------------------------------------------
// Heads: loc = new @ locW + locb, cls = new @ clsW + clsb, written straight
// into d_out at layout [B,16,6,4] (loc) then [B,16,6,4] (cls).
// ---------------------------------------------------------------------------
__global__ void heads(const _Float16* __restrict__ nw,
                      const float* __restrict__ locW, const float* __restrict__ locb,
                      const float* __restrict__ clsW, const float* __restrict__ clsb,
                      float* __restrict__ out, int t)
{
    int idx = blockIdx.x * blockDim.x + threadIdx.x;        // [0, B*6*8)
    if (idx >= B_SZ * N_OBJ * 8) return;
    int row = idx >> 3;
    int c   = idx & 7;                                      // 0-3 loc, 4-7 cls
    int b   = row / N_OBJ;
    int n   = row % N_OBJ;
    int cc  = c & 3;
    const float* W  = (c < 4) ? locW : clsW;
    float s         = (c < 4) ? locb[cc] : clsb[cc];
    const _Float16* x = nw + (size_t)row * D_DIM;
#pragma unroll 8
    for (int k = 0; k < D_DIM; ++k)
        s += (float)x[k] * W[k * 4 + cc];
    size_t base = (c < 4) ? 0 : (size_t)B_SZ * T_STEP * N_OBJ * 4;
    out[base + (size_t)b * (T_STEP * N_OBJ * 4) + (size_t)t * (N_OBJ * 4) + n * 4 + cc] = s;
}

// ---------------------------------------------------------------------------
extern "C" void kernel_launch(void* const* d_in, const int* in_sizes, int n_in,
                              void* d_out, int out_size, void* d_ws, size_t ws_size,
                              hipStream_t stream) {
    (void)in_sizes; (void)n_in; (void)out_size; (void)ws_size;
    const float* node  = (const float*)d_in[0];
    const float* black = (const float*)d_in[1];
    // d_in[2]=Rs, d_in[3]=Rr : fixed incidence, computed analytically.
    const float* rW1 = (const float*)d_in[4];  const float* rb1 = (const float*)d_in[5];
    const float* rW2 = (const float*)d_in[6];  const float* rb2 = (const float*)d_in[7];
    const float* rW3 = (const float*)d_in[8];  const float* rb3 = (const float*)d_in[9];
    const float* rW4 = (const float*)d_in[10]; const float* rb4 = (const float*)d_in[11];
    const float* oW1 = (const float*)d_in[12]; const float* ob1 = (const float*)d_in[13];
    const float* oW2 = (const float*)d_in[14]; const float* ob2 = (const float*)d_in[15];
    const float* locW = (const float*)d_in[16]; const float* locb = (const float*)d_in[17];
    const float* clsW = (const float*)d_in[18]; const float* clsb = (const float*)d_in[19];
    float* out = (float*)d_out;

    // ---- workspace carve (fp16 halves), ~42 MB total -----------------------
    _Float16* base = (_Float16*)d_ws;
    size_t off = 0;
    auto alloc = [&](size_t n) { _Float16* p = base + off; off += (n + 127) & ~(size_t)127; return p; };
    _Float16* wt_r1 = alloc(256 * 256);
    _Float16* wt_r2 = alloc(256 * 256);
    _Float16* wt_r3 = alloc(128 * 256);
    _Float16* wt_r4 = alloc(128 * 128);
    _Float16* wt_o1 = alloc(256 * 384);
    _Float16* wt_o2 = alloc(128 * 256);
    _Float16* nd16  = alloc((size_t)B_SZ * N_OBJ * D_DIM);     // node state (fp16)
    _Float16* blk16 = alloc((size_t)B_SZ * D_DIM);
    _Float16* bufA  = alloc((size_t)B_SZ * N_REL * 256);       // 15.7 MB
    _Float16* bufB  = alloc((size_t)B_SZ * N_REL * 256);       // 15.7 MB
    _Float16* u16   = alloc((size_t)B_SZ * N_OBJ * 384);
    _Float16* ho16  = alloc((size_t)B_SZ * N_OBJ * 256);

    // ---- one-time prep: weight transpose+cvt, input cvt --------------------
    auto tc = [&](const float* W, _Float16* Wt, int K, int N) {
        transpose_cvt<<<(K * N + 255) / 256, 256, 0, stream>>>(W, Wt, K, N);
    };
    tc(rW1, wt_r1, 256, 256);  tc(rW2, wt_r2, 256, 256);
    tc(rW3, wt_r3, 256, 128);  tc(rW4, wt_r4, 128, 128);
    tc(oW1, wt_o1, 384, 256);  tc(oW2, wt_o2, 256, 128);

    const int ndElems = B_SZ * N_OBJ * D_DIM;
    cvt_f32_f16<<<(ndElems + 255) / 256, 256, 0, stream>>>(node, nd16, ndElems);
    cvt_f32_f16<<<(B_SZ * D_DIM + 255) / 256, 256, 0, stream>>>(black, blk16, B_SZ * D_DIM);

    // ---- T sequential interaction-network steps ----------------------------
    const int MR = B_SZ * N_REL;   // 30720, multiple of 256
    const int MO = B_SZ * N_OBJ;   // 6144,  multiple of 256
    for (int t = 0; t < T_STEP; ++t) {
        gather_e<<<(B_SZ * N_REL * 32 + 255) / 256, 256, 0, stream>>>(nd16, bufA);

        gemm_bias_act<256, 256, true ><<<dim3(MR / 256, 4), 256, 0, stream>>>(bufA, wt_r1, rb1, bufB);
        gemm_bias_act<256, 256, true ><<<dim3(MR / 256, 4), 256, 0, stream>>>(bufB, wt_r2, rb2, bufA);
        gemm_bias_act<256, 128, true ><<<dim3(MR / 256, 2), 256, 0, stream>>>(bufA, wt_r3, rb3, bufB);
        gemm_bias_act<128, 128, true ><<<dim3(MR / 256, 2), 256, 0, stream>>>(bufB, wt_r4, rb4, bufA);

        build_u<<<MO, 128, 0, stream>>>(nd16, bufA, blk16, u16);

        gemm_bias_act<384, 256, true ><<<dim3(MO / 256, 4), 256, 0, stream>>>(u16, wt_o1, ob1, ho16);
        gemm_bias_act<256, 128, false><<<dim3(MO / 256, 2), 256, 0, stream>>>(ho16, wt_o2, ob2, nd16);

        heads<<<(B_SZ * N_OBJ * 8 + 255) / 256, 256, 0, stream>>>(
            nd16, locW, locb, clsW, clsb, out, t);
    }

    // ---- final node state -> fp32 'last' output ----------------------------
    float* lastOut = out + (size_t)2 * B_SZ * T_STEP * N_OBJ * 4;   // after loc+cls
    cvt_f16_f32<<<(ndElems + 255) / 256, 256, 0, stream>>>(nd16, lastOut, ndElems);
}
